// QuantLinear_45672682226460
// MI455X (gfx1250) — compile-verified, matrix operand order checked by reading, not run on hardware
//
#include <hip/hip_runtime.h>
#include <stdint.h>

#define IN_F   4096
#define OUT_F  11008
#define M_ROWS 8
#define K_CHUNK 512          // K per split
#define N_TILES (OUT_F / 16) // 688

typedef __attribute__((ext_vector_type(16))) _Float16 v16h;
typedef __attribute__((ext_vector_type(8)))  float    v8f;

union V16H {
    v16h     v;
    uint32_t u[8];
};

// ---------------------------------------------------------------------------
// Kernel 1: convert x (8 x 4096 fp32) -> fp16 in workspace with a per-8-group
// K permutation, and compute per-(row, K-chunk) rowsums.
//
// K permutation: within each group of 8 consecutive K, position p holds the
// element with actual offset a(p) = (p>>1) + (p&1)*4. This makes the B-side
// nibble unpack a single shift + fused and-or per fp16 pair:
//   pair j of a qweight dword = nibbles (j, j+4) = ((q>>4j) & 0x000F000F).
// One wave per (row, chunk): 8 rows x 8 chunks = 64 blocks of 32 threads.
// ---------------------------------------------------------------------------
__global__ void xcvt_kernel(const float* __restrict__ x,
                            _Float16* __restrict__ xh,
                            float* __restrict__ rowsum) {
    const int row   = blockIdx.x & 7;
    const int chunk = blockIdx.x >> 3;
    const int t     = threadIdx.x;                  // 0..31
    const int k     = chunk * K_CHUNK + t * 16;     // 16 elements per lane

    const float* src = x + row * IN_F + k;
    union { _Float16 h[16]; uint4 q[2]; } buf;

    float s = 0.0f;
#pragma unroll
    for (int i = 0; i < 16; ++i) {
        const int g = i >> 3;                 // which 8-group
        const int p = i & 7;                  // position within group
        const int a = (p >> 1) + ((p & 1) << 2);   // actual K offset
        float f  = src[g * 8 + a];
        float fs = src[i];                    // rowsum over natural order
        s += fs;
        buf.h[i] = (_Float16)f;
    }
    uint4* dst = reinterpret_cast<uint4*>(xh + row * IN_F + k);
    dst[0] = buf.q[0];
    dst[1] = buf.q[1];

    // wave32 reduction of the chunk rowsum
#pragma unroll
    for (int off = 16; off > 0; off >>= 1)
        s += __shfl_down(s, off);
    if (t == 0)
        rowsum[chunk * M_ROWS + row] = s;
}

// ---------------------------------------------------------------------------
// Kernel 2: out[m][n] = bias[n]  (split-K partials are atomically added later)
// ---------------------------------------------------------------------------
__global__ void init_out_kernel(float* __restrict__ out,
                                const float* __restrict__ bias) {
    int i = blockIdx.x * blockDim.x + threadIdx.x;
    if (i < M_ROWS * OUT_F)
        out[i] = bias[i % OUT_F];
}

// ---------------------------------------------------------------------------
// Kernel 3: main GPTQ WMMA kernel.
// grid = (N_TILES/4, 8 K-splits), block = 128 (4 independent waves).
// Each wave: 16 output columns x 512 K -> 16 x v_wmma_f32_16x16x32_f16.
//
// B fragment (K=32 x N=16 fp16): lane l holds column (l&15); dword q0 of
// qweight row r supplies VGPRs 0..3 as pairs (nibble j, nibble j+4) via
//   b.u[j] = ((q0 >> 4j) & 0x000F000F) | 0x64006400      (fp16 = 1024 + w)
// and q1 (two packed rows down, K+16) supplies VGPRs 4..7. The matching K
// permutation was baked into the fp16 x buffer by xcvt_kernel.
// The 1024 offset and the per-column zero-point are removed after the K loop:
//   partial_out = scale[n] * ( S - (1024 + zq[n] + 1) * rowsum[m] )
// A fragment: lane l holds x row (l&15); rows 8..15 are don't-cares (their
// products only reach D rows 8..15 = lanes 16..31, which are discarded), so
// those lanes simply re-load rows 0..7.
// ---------------------------------------------------------------------------
__global__ void __launch_bounds__(128, 1)
gptq_wmma_kernel(const int* __restrict__ qweight,
                 const int* __restrict__ qzeros,
                 const float* __restrict__ scales,
                 const _Float16* __restrict__ xh,
                 const float* __restrict__ rowsum,
                 float* __restrict__ out) {
    const int lane  = threadIdx.x & 31;
    const int wave  = threadIdx.x >> 5;
    const int ntile = blockIdx.x * 4 + wave;     // 0..687
    const int chunk = blockIdx.y;                // 0..7 K split
    const int nlo   = lane & 15;
    const int half  = lane >> 4;                 // lane half selects K sub-block
    const int col   = ntile * 16 + nlo;
    const int ra    = lane & 7;                  // x row (dup for lanes 8..15)

    // per-column zero-point nibble and scale
    const uint32_t qz = (uint32_t)qzeros[col >> 3];
    const float    zq = (float)((qz >> ((col & 7) * 4)) & 0xFu);
    const float    sc = scales[col];

    const int kb0 = chunk * (K_CHUNK / 8);       // packed-row base (chunk*64)
    const _Float16* xrow = xh + ra * IN_F + chunk * K_CHUNK + half * 8;
    const int*      qcol = qweight + col;        // column stride OUT_F dwords

    v8f c = {};                                  // fp32 accumulators

#pragma unroll 4
    for (int it = 0; it < K_CHUNK / 32; ++it) {
        // ---- A fragment: two contiguous 16B loads of (permuted) fp16 x
        V16H a;
        const uint4 a0 = *reinterpret_cast<const uint4*>(xrow + it * 32);
        const uint4 a1 = *reinterpret_cast<const uint4*>(xrow + it * 32 + 16);
        a.u[0] = a0.x; a.u[1] = a0.y; a.u[2] = a0.z; a.u[3] = a0.w;
        a.u[4] = a1.x; a.u[5] = a1.y; a.u[6] = a1.z; a.u[7] = a1.w;

        // ---- B fragment: two coalesced dwords of packed nibbles per lane
        const int r0 = kb0 + it * 4 + half;
        const uint32_t q0 = (uint32_t)qcol[r0 * OUT_F];
        const uint32_t q1 = (uint32_t)qcol[(r0 + 2) * OUT_F];

        V16H b;
#pragma unroll
        for (int j = 0; j < 4; ++j) {
            // pair = nibbles (j, j+4): one shift + one fused and-or (bitop3)
            b.u[j]     = ((q0 >> (4 * j)) & 0x000F000Fu) | 0x64006400u;
            b.u[4 + j] = ((q1 >> (4 * j)) & 0x000F000Fu) | 0x64006400u;
        }

        c = __builtin_amdgcn_wmma_f32_16x16x32_f16(
                /*neg_a=*/false, a.v, /*neg_b=*/false, b.v,
                /*c_mod=*/(short)0, c, /*reuse_a=*/false, /*reuse_b=*/false);
    }

    // lanes 0..15 hold rows M = v (0..7); lanes 16..31 hold padding rows.
    if (lane < 16) {
        const float coef = 1025.0f + zq;         // 1024 offset + (zq + 1)
#pragma unroll
        for (int v = 0; v < 8; ++v) {
            float res = sc * (c[v] - coef * rowsum[chunk * M_ROWS + v]);
            atomicAdd(out + v * OUT_F + col, res);
        }
    }
}

// ---------------------------------------------------------------------------
// inputs (setup_inputs order): x, qweight, qzeros, scales, bias
// ---------------------------------------------------------------------------
extern "C" void kernel_launch(void* const* d_in, const int* in_sizes, int n_in,
                              void* d_out, int out_size, void* d_ws, size_t ws_size,
                              hipStream_t stream) {
    const float* x       = (const float*)d_in[0];
    const int*   qweight = (const int*)  d_in[1];
    const int*   qzeros  = (const int*)  d_in[2];
    const float* scales  = (const float*)d_in[3];
    const float* bias    = (const float*)d_in[4];
    float*       out     = (float*)d_out;

    _Float16* xh     = (_Float16*)d_ws;                                   // 64 KB
    float*    rowsum = (float*)((char*)d_ws + M_ROWS * IN_F * sizeof(_Float16));

    // 1) x -> permuted fp16 + per-chunk rowsums (recomputed every call)
    xcvt_kernel<<<64, 32, 0, stream>>>(x, xh, rowsum);

    // 2) out = bias broadcast (split-K partials accumulate on top)
    init_out_kernel<<<(M_ROWS * OUT_F + 255) / 256, 256, 0, stream>>>(out, bias);

    // 3) main kernel: 688 N-tiles x 8 K-splits = 5504 waves
    gptq_wmma_kernel<<<dim3(N_TILES / 4, 8), 128, 0, stream>>>(
        qweight, qzeros, scales, xh, rowsum, out);
}
